// p_model_29162827939891
// MI455X (gfx1250) — compile-verified, matrix operand order checked by reading, not run on hardware
//
#include <hip/hip_runtime.h>
#include <hip/hip_bf16.h>

#define HH   20      // hidden size
#define KP   32      // padded K for WMMA
#define NG   80      // 4*H gates
#define TILE 16      // batch rows per workgroup
#define BATCH 1024
#define LSEQ  2048
#define OUTC  2047   // output columns
#define PRED  128
#define WARM  1024   // stepp_warm steps
#define POUT  127    // stepp_out steps

typedef __attribute__((ext_vector_type(8)))  _Float16 v8h;
typedef __attribute__((ext_vector_type(16))) _Float16 v16h;
typedef __attribute__((ext_vector_type(8)))  float    v8f;

struct __align__(16) Smem {
  // weights as f16 row-major [rows][KP], zero padded K=HH..KP
  _Float16 Wg0 [NG*KP];   // Whh_g0
  _Float16 Wr1 [NG*KP];   // Whh_r1
  _Float16 Wir2[NG*KP];   // Wih_r2
  _Float16 Whr2[NG*KP];   // Whh_r2
  _Float16 Wp1 [NG*KP];   // Whh_p1
  _Float16 Wip2[NG*KP];   // Wih_p2
  _Float16 Whp2[NG*KP];   // Whh_p2
  _Float16 W1y [64*KP];
  _Float16 W1x [64*KP];
  _Float16 W1p [64*KP];
  // recurrent h state in A-fragment-friendly layout [TILE][KP] f16
  _Float16 hg0[TILE*KP];
  _Float16 hr1[TILE*KP];
  _Float16 hr2[TILE*KP];
  _Float16 hp1[TILE*KP];
  _Float16 hp2[TILE*KP];
  float gates[TILE*NG];      // [batch][gate]
  float zbuf [64*TILE];      // transposed [col][batch]
  float bsum_g0[NG], bsum_r1[NG], bsum_r2[NG], bsum_p1[NG], bsum_p2[NG];
  float wih0_g0[NG], wih0_r1[NG], wih1_r1[NG], wih0_p1[NG];
  float b1y[64], W2y[64], b1x[64], W2x[64], b1p[64], W2p[64];
  float xbuf[TILE], yprev[TILE], cum[TILE];
  float predcum[TILE*POUT];
  float b2y, b2x, b2p;
};

struct KParams {
  const float* x;
  const float* Wih_g0; const float* Whh_g0; const float* bih_g0; const float* bhh_g0;
  const float* Wih_r1; const float* Whh_r1; const float* bih_r1; const float* bhh_r1;
  const float* Wih_r2; const float* Whh_r2; const float* bih_r2; const float* bhh_r2;
  const float* Wih_p1; const float* Whh_p1; const float* bih_p1; const float* bhh_p1;
  const float* Wih_p2; const float* Whh_p2; const float* bih_p2; const float* bhh_p2;
  const float* W1x; const float* b1x; const float* W2x; const float* b2x;
  const float* W1y; const float* b1y; const float* W2y; const float* b2y;
  const float* W1p; const float* b1p; const float* W2p; const float* b2p;
  float* out_output;  // (B, 2047)
  float* out_g0;      // (B, 1)
  float* out_pred;    // (B, 128)
};

__device__ __forceinline__ float sigf(float x) { return 1.0f / (1.0f + __expf(-x)); }

// Load a 16x32 f16 WMMA A/B fragment from row-major LDS [16][KP] f16.
// 16-bit A layout: lanes 0-15 hold row (lane), K0..7 + K16..23;
// lanes 16-31 hold row (lane-16), K8..15 + K24..31. (B mirrored with N per lane.)
__device__ __forceinline__ v16h ld_frag(const _Float16* base, int lane) {
  const int row = lane & 15;
  const int off = (lane & 16) ? 8 : 0;
  const _Float16* p = base + row * KP + off;
  v8h lo = *(const v8h*)(p);
  v8h hi = *(const v8h*)(p + 16);
  return __builtin_shufflevector(lo, hi, 0,1,2,3,4,5,6,7,8,9,10,11,12,13,14,15);
}

__device__ __forceinline__ v8f wmma32(v16h a, v16h b, v8f c) {
  return __builtin_amdgcn_wmma_f32_16x16x32_f16(false, a, false, b, (short)0, c, false, false);
}

// gates[16x80] = hA(16x20) @ Whh.T + (optional) inA(16x20) @ Wih.T   (5 N-tiles)
__device__ __forceinline__ void gate_gemm(float* gates, const _Float16* hA,
                                          const _Float16* Whh,
                                          const _Float16* inA, const _Float16* Wih,
                                          int lane) {
  v16h a = ld_frag(hA, lane);
  v16h a2 = {};
  if (inA) a2 = ld_frag(inA, lane);
  const int col0 = lane & 15;
  const int rb   = (lane & 16) ? 8 : 0;
  #pragma unroll
  for (int n = 0; n < 5; ++n) {
    v8f acc = {};
    if (inA) acc = wmma32(a2, ld_frag(Wih + n * 16 * KP, lane), acc);
    acc = wmma32(a, ld_frag(Whh + n * 16 * KP, lane), acc);
    const int col = n * 16 + col0;
    #pragma unroll
    for (int r = 0; r < 8; ++r) gates[(rb + r) * NG + col] = acc[r];
  }
}

// LSTM elementwise: lane owns batch b=lane&15, hidden (lane<16 ? 0..9 : 10..19)
__device__ __forceinline__ void cell_elem(const float* gates, float* c, _Float16* hA,
                                          const float* bsum,
                                          const float* wih0, const float* wih1,
                                          const float* x1v, const float* x2v, int lane) {
  const int b  = lane & 15;
  const int hb = (lane & 16) ? 10 : 0;
  const float x1 = x1v ? x1v[b] : 0.0f;
  const float x2 = x2v ? x2v[b] : 0.0f;
  #pragma unroll
  for (int j = 0; j < 10; ++j) {
    const int hd = hb + j;
    float iv = gates[b * NG + hd]      + bsum[hd];
    float fv = gates[b * NG + 20 + hd] + bsum[20 + hd];
    float gv = gates[b * NG + 40 + hd] + bsum[40 + hd];
    float ov = gates[b * NG + 60 + hd] + bsum[60 + hd];
    if (wih0) { iv += x1 * wih0[hd]; fv += x1 * wih0[20 + hd];
                gv += x1 * wih0[40 + hd]; ov += x1 * wih0[60 + hd]; }
    if (wih1) { iv += x2 * wih1[hd]; fv += x2 * wih1[20 + hd];
                gv += x2 * wih1[40 + hd]; ov += x2 * wih1[60 + hd]; }
    const float cc = sigf(fv) * c[j] + sigf(iv) * tanhf(gv);
    const float hh = sigf(ov) * tanhf(cc);
    c[j] = cc;
    hA[b * KP + hd] = (_Float16)hh;
  }
}

// zbuf[col][batch] = act(hA(16x20) @ W1.T(20x64) + b1)   (4 N-tiles)
__device__ __forceinline__ void head_gemm(float* zbuf, const _Float16* hA,
                                          const _Float16* W1, const float* b1,
                                          bool relu, int lane) {
  v16h a = ld_frag(hA, lane);
  const int col0 = lane & 15;
  const int rb   = (lane & 16) ? 8 : 0;
  #pragma unroll
  for (int n = 0; n < 4; ++n) {
    v8f acc = {};
    acc = wmma32(a, ld_frag(W1 + n * 16 * KP, lane), acc);
    const int col = n * 16 + col0;
    #pragma unroll
    for (int r = 0; r < 8; ++r) {
      float v = acc[r] + b1[col];
      if (relu) v = fmaxf(v, 0.0f);
      zbuf[col * TILE + (rb + r)] = v;
    }
  }
}

__device__ __forceinline__ void cvt_w(_Float16* dst, const float* src, int rows, int tid) {
  for (int i = tid; i < rows * KP; i += 32) {
    const int r = i >> 5, k = i & 31;
    dst[i] = (_Float16)((k < HH) ? src[r * HH + k] : 0.0f);
  }
}

__global__ __launch_bounds__(32) void lstm_fused_kernel(KParams P) {
  extern __shared__ char smem_raw[];
  Smem& S = *(Smem*)smem_raw;
  const int lane = threadIdx.x;
  const int row0 = blockIdx.x * TILE;

  // ---- one-time LDS setup -------------------------------------------------
  cvt_w(S.Wg0,  P.Whh_g0, NG, lane);
  cvt_w(S.Wr1,  P.Whh_r1, NG, lane);
  cvt_w(S.Wir2, P.Wih_r2, NG, lane);
  cvt_w(S.Whr2, P.Whh_r2, NG, lane);
  cvt_w(S.Wp1,  P.Whh_p1, NG, lane);
  cvt_w(S.Wip2, P.Wih_p2, NG, lane);
  cvt_w(S.Whp2, P.Whh_p2, NG, lane);
  cvt_w(S.W1y,  P.W1y, 64, lane);
  cvt_w(S.W1x,  P.W1x, 64, lane);
  cvt_w(S.W1p,  P.W1p, 64, lane);
  for (int g = lane; g < NG; g += 32) {
    S.bsum_g0[g] = P.bih_g0[g] + P.bhh_g0[g];
    S.bsum_r1[g] = P.bih_r1[g] + P.bhh_r1[g];
    S.bsum_r2[g] = P.bih_r2[g] + P.bhh_r2[g];
    S.bsum_p1[g] = P.bih_p1[g] + P.bhh_p1[g];
    S.bsum_p2[g] = P.bih_p2[g] + P.bhh_p2[g];
    S.wih0_g0[g] = P.Wih_g0[g];
    S.wih0_r1[g] = P.Wih_r1[g * 2 + 0];
    S.wih1_r1[g] = P.Wih_r1[g * 2 + 1];
    S.wih0_p1[g] = P.Wih_p1[g];
  }
  for (int n = lane; n < 64; n += 32) {
    S.b1y[n] = P.b1y[n]; S.W2y[n] = P.W2y[n];
    S.b1x[n] = P.b1x[n]; S.W2x[n] = P.W2x[n];
    S.b1p[n] = P.b1p[n]; S.W2p[n] = P.W2p[n];
  }
  if (lane == 0) { S.b2y = P.b2y[0]; S.b2x = P.b2x[0]; S.b2p = P.b2p[0]; }
  for (int i = lane; i < TILE * KP; i += 32) {
    S.hg0[i] = (_Float16)0.0f; S.hr1[i] = (_Float16)0.0f; S.hr2[i] = (_Float16)0.0f;
    S.hp1[i] = (_Float16)0.0f; S.hp2[i] = (_Float16)0.0f;
  }
  __syncthreads();

  const float* xg = P.x;
  float cg0[10], cr1[10], cr2[10], cp1[10], cp2[10];
  #pragma unroll
  for (int j = 0; j < 10; ++j) { cg0[j]=0.f; cr1[j]=0.f; cr2[j]=0.f; cp1[j]=0.f; cp2[j]=0.f; }

  // ---- phase 0: g0 LSTM over x[:, 1:]  (2047 steps) -----------------------
  #pragma unroll 1
  for (int t = 0; t < LSEQ - 1; ++t) {
    if (lane < TILE) {
      S.xbuf[lane] = xg[(size_t)(row0 + lane) * LSEQ + 1 + t];
      __builtin_prefetch(&xg[(size_t)(row0 + lane) * LSEQ + 1 + t + 32], 0, 1);
    }
    __syncthreads();
    gate_gemm(S.gates, S.hg0, S.Wg0, nullptr, nullptr, lane);
    __syncthreads();
    cell_elem(S.gates, cg0, S.hg0, S.bsum_g0, S.wih0_g0, nullptr, S.xbuf, nullptr, lane);
    __syncthreads();
  }
  // g0 head: (h0 @ W1y.T + b1y) @ W2y.T + b2y   (no relu)
  head_gemm(S.zbuf, S.hg0, S.W1y, S.b1y, false, lane);
  __syncthreads();
  float cumreg = 0.0f, paccreg = 0.0f;
  if (lane < TILE) {
    float acc = S.b2y;
    #pragma unroll 8
    for (int k = 0; k < 64; ++k) acc += S.zbuf[k * TILE + lane] * S.W2y[k];
    cumreg = acc;
    P.out_g0[row0 + lane] = acc;
    P.out_output[(size_t)(row0 + lane) * OUTC + 0] = acc;
    S.yprev[lane] = acc;
    S.cum[lane]   = acc;
  }
  __syncthreads();

  // ---- fused phase 1 (r1/r2 + x-head) + phase 2 (p1/p2 + p-head) ----------
  #pragma unroll 1
  for (int t = 0; t < LSEQ - 2; ++t) {
    if (lane < TILE) S.xbuf[lane] = xg[(size_t)(row0 + lane) * LSEQ + 1 + t];
    __syncthreads();
    // r1: input = [x_t, y_prev]
    gate_gemm(S.gates, S.hr1, S.Wr1, nullptr, nullptr, lane);
    __syncthreads();
    cell_elem(S.gates, cr1, S.hr1, S.bsum_r1, S.wih0_r1, S.wih1_r1, S.xbuf, S.yprev, lane);
    __syncthreads();
    // r2: input = h1 (via WMMA)
    gate_gemm(S.gates, S.hr2, S.Whr2, S.hr1, S.Wir2, lane);
    __syncthreads();
    cell_elem(S.gates, cr2, S.hr2, S.bsum_r2, nullptr, nullptr, nullptr, nullptr, lane);
    __syncthreads();
    // y1 = relu(h2 @ W1x.T + b1x) @ W2x.T + b2x ; cum += y1
    head_gemm(S.zbuf, S.hr2, S.W1x, S.b1x, true, lane);
    __syncthreads();
    if (lane < TILE) {
      float acc = S.b2x;
      #pragma unroll 8
      for (int k = 0; k < 64; ++k) acc += S.zbuf[k * TILE + lane] * S.W2x[k];
      cumreg += acc;
      S.yprev[lane] = acc;
      S.cum[lane]   = cumreg;
      P.out_output[(size_t)(row0 + lane) * OUTC + 1 + t] = cumreg;
    }
    __syncthreads();
    // phase-2 step consumes output column (t+1) == cum just produced
    if (t < WARM + POUT) {
      gate_gemm(S.gates, S.hp1, S.Wp1, nullptr, nullptr, lane);
      __syncthreads();
      cell_elem(S.gates, cp1, S.hp1, S.bsum_p1, S.wih0_p1, nullptr, S.cum, nullptr, lane);
      __syncthreads();
      gate_gemm(S.gates, S.hp2, S.Whp2, S.hp1, S.Wip2, lane);
      __syncthreads();
      cell_elem(S.gates, cp2, S.hp2, S.bsum_p2, nullptr, nullptr, nullptr, nullptr, lane);
      __syncthreads();
      if (t >= WARM) {
        head_gemm(S.zbuf, S.hp2, S.W1p, S.b1p, true, lane);
        __syncthreads();
        if (lane < TILE) {
          float acc = S.b2p;
          #pragma unroll 8
          for (int k = 0; k < 64; ++k) acc += S.zbuf[k * TILE + lane] * S.W2p[k];
          paccreg += acc;
          S.predcum[lane * POUT + (t - WARM)] = paccreg;
        }
        __syncthreads();
      }
    }
  }

  // ---- prediction epilogue: p0 = output[:, -1]; pred = [p0, p0+cumsum] ----
  if (lane < TILE) {
    const float p0 = cumreg;
    float* pr = P.out_pred + (size_t)(row0 + lane) * PRED;
    pr[0] = p0;
    #pragma unroll 1
    for (int j = 0; j < POUT; ++j) pr[1 + j] = p0 + S.predcum[lane * POUT + j];
  }
}

extern "C" void kernel_launch(void* const* d_in, const int* in_sizes, int n_in,
                              void* d_out, int out_size, void* d_ws, size_t ws_size,
                              hipStream_t stream) {
  (void)in_sizes; (void)n_in; (void)d_ws; (void)ws_size; (void)out_size;
  KParams P;
  P.x      = (const float*)d_in[0];
  P.Wih_g0 = (const float*)d_in[1];  P.Whh_g0 = (const float*)d_in[2];
  P.bih_g0 = (const float*)d_in[3];  P.bhh_g0 = (const float*)d_in[4];
  P.Wih_r1 = (const float*)d_in[5];  P.Whh_r1 = (const float*)d_in[6];
  P.bih_r1 = (const float*)d_in[7];  P.bhh_r1 = (const float*)d_in[8];
  P.Wih_r2 = (const float*)d_in[9];  P.Whh_r2 = (const float*)d_in[10];
  P.bih_r2 = (const float*)d_in[11]; P.bhh_r2 = (const float*)d_in[12];
  P.Wih_p1 = (const float*)d_in[13]; P.Whh_p1 = (const float*)d_in[14];
  P.bih_p1 = (const float*)d_in[15]; P.bhh_p1 = (const float*)d_in[16];
  P.Wih_p2 = (const float*)d_in[17]; P.Whh_p2 = (const float*)d_in[18];
  P.bih_p2 = (const float*)d_in[19]; P.bhh_p2 = (const float*)d_in[20];
  P.W1x = (const float*)d_in[21]; P.b1x = (const float*)d_in[22];
  P.W2x = (const float*)d_in[23]; P.b2x = (const float*)d_in[24];
  P.W1y = (const float*)d_in[25]; P.b1y = (const float*)d_in[26];
  P.W2y = (const float*)d_in[27]; P.b2y = (const float*)d_in[28];
  P.W1p = (const float*)d_in[29]; P.b1p = (const float*)d_in[30];
  P.W2p = (const float*)d_in[31]; P.b2p = (const float*)d_in[32];

  float* out = (float*)d_out;
  P.out_output = out;                                      // 1024*2047
  P.out_g0     = out + (size_t)BATCH * OUTC;               // 1024
  P.out_pred   = out + (size_t)BATCH * OUTC + BATCH;       // 1024*128

  const dim3 grid(BATCH / TILE);   // 64 workgroups, each owns 16 batch rows
  const dim3 block(32);            // one wave32 per workgroup (WMMA granularity)
  lstm_fused_kernel<<<grid, block, sizeof(Smem), stream>>>(P);
}
